// DecoderBlock_6167573037770
// MI455X (gfx1250) — compile-verified, matrix operand order checked by reading, not run on hardware
//
#include <hip/hip_runtime.h>
#include <hip/hip_bf16.h>
#include <math.h>

// ---------------------------------------------------------------------------
// Decoder block on MI455X (gfx1250): bf16 WMMA everywhere, f32 accumulate.
// B=4, S=2048, D=1024, H=16, dh=64, Dff=4096, M=B*S=8192.
// ---------------------------------------------------------------------------

typedef __hip_bfloat16 bf16_t;
typedef __attribute__((ext_vector_type(16))) __bf16 v16bf;
typedef __attribute__((ext_vector_type(8)))  float  v8f;

#define B_   4
#define S_   2048
#define D_   1024
#define H_   16
#define DH_  64
#define DFF_ 4096
#define M_   (B_ * S_)

union FragU { uint4 q[2]; v16bf v; };

// 16-bit A-matrix fragment (ISA 7.12.2): lanes 0-15 hold row M=lane,
// K = {k0..k0+7} in VGPR0-3 and {k0+16..k0+23} in VGPR4-7; lanes 16-31 hold
// the same rows with K offset by 8. B-fragments use the identical pattern on
// a [N,K] (pre-transposed) operand, so one helper serves both.
__device__ __forceinline__ v16bf load_frag(const bf16_t* base, int ld,
                                           int row0, int k0, int lane) {
  int r  = row0 + (lane & 15);
  int kh = (lane & 16) ? 8 : 0;
  const bf16_t* p = base + (size_t)r * ld + (size_t)(k0 + kh);
  FragU f;
  f.q[0] = *(const uint4*)(p);        // K = kh .. kh+7
  f.q[1] = *(const uint4*)(p + 16);   // K = kh+16 .. kh+23
  return f.v;
}

__device__ __forceinline__ v8f wmma_bf16(v16bf a, v16bf b, v8f c) {
  return __builtin_amdgcn_wmma_f32_16x16x32_bf16(false, a, false, b,
                                                 (short)0, c, false, false);
}

// ---------------------------------------------------------------------------
// LayerNorm (fp32 stats) -> bf16 activations. One row per block, 256 threads.
// ---------------------------------------------------------------------------
__global__ __launch_bounds__(256) void ln_bf16_kernel(
    const float* __restrict__ x, const float* __restrict__ alpha,
    const float* __restrict__ beta, bf16_t* __restrict__ y) {
  __shared__ float sh[16];
  int row = blockIdx.x;
  const float* xr = x + (size_t)row * D_;
  float v[4], s1 = 0.f, s2 = 0.f;
#pragma unroll
  for (int j = 0; j < 4; ++j) {
    v[j] = xr[threadIdx.x + j * 256];
    s1 += v[j];
    s2 += v[j] * v[j];
  }
#pragma unroll
  for (int m = 16; m >= 1; m >>= 1) {
    s1 += __shfl_xor(s1, m, 32);
    s2 += __shfl_xor(s2, m, 32);
  }
  int wid = threadIdx.x >> 5, lane = threadIdx.x & 31;
  if (lane == 0) { sh[wid] = s1; sh[8 + wid] = s2; }
  __syncthreads();
  s1 = 0.f; s2 = 0.f;
#pragma unroll
  for (int w = 0; w < 8; ++w) { s1 += sh[w]; s2 += sh[8 + w]; }
  float mean = s1 * (1.0f / D_);
  float var  = s2 * (1.0f / D_) - mean * mean;
  float rstd = rsqrtf(var + 1e-5f);
  bf16_t* yr = y + (size_t)row * D_;
#pragma unroll
  for (int j = 0; j < 4; ++j) {
    int c = threadIdx.x + j * 256;
    yr[c] = __float2bfloat16(alpha[c] * (v[j] - mean) * rstd + beta[c]);
  }
}

// ---------------------------------------------------------------------------
// Transpose-cast fp32 [K,N] -> bf16 [N,K] (fragment-friendly weight layout).
// ---------------------------------------------------------------------------
__global__ __launch_bounds__(256) void cvt_t_kernel(
    const float* __restrict__ W, bf16_t* __restrict__ Wt, int K, int N) {
  size_t idx = (size_t)blockIdx.x * 256 + threadIdx.x;
  if (idx >= (size_t)K * N) return;
  int n = (int)(idx % N);
  int k = (int)(idx / N);
  Wt[(size_t)n * K + k] = __float2bfloat16(W[idx]);
}

// ---------------------------------------------------------------------------
// WMMA GEMM: C[M,N] = A[M,K](bf16,row-major) * Wt[N,K](bf16)^T, epilogues:
//   EP_F32      : f32 out (+bias)(+residual)
//   EP_GELU_BF16: bf16 out, exact GELU(acc + bias)
//   EP_Q / EP_K : scatter bf16 to [B,H,S,dh]  (Q pre-scaled by 0.125)
//   EP_VT       : scatter bf16 to [B,H,dh,S]  (transposed V)
// 256 threads = 8 waves (4x2), wave tile 32x64 (2x4 accums -> 8 WMMAs per
// 6 fragment loads per K-step), block tile 128x128. K-loop unrolled 2x so
// two K-steps of independent b128 loads stay in flight against the XDL pipe.
// ---------------------------------------------------------------------------
enum { EP_F32 = 0, EP_GELU_BF16 = 1, EP_Q = 2, EP_K = 3, EP_VT = 4 };

template <int MODE>
__global__ __launch_bounds__(256, 1) void gemm_wmma_kernel(
    const bf16_t* __restrict__ A, const bf16_t* __restrict__ Wt,
    int M, int N, int K,
    const float* __restrict__ bias, const float* __restrict__ resid,
    float* __restrict__ outF, bf16_t* __restrict__ outB) {
  int lane = threadIdx.x & 31;
  int wid  = threadIdx.x >> 5;
  int m0 = blockIdx.x * 128 + (wid >> 1) * 32;
  int n0 = blockIdx.y * 128 + (wid & 1) * 64;

  v8f acc[2][4];
#pragma unroll
  for (int i = 0; i < 2; ++i)
#pragma unroll
    for (int j = 0; j < 4; ++j)
#pragma unroll
      for (int z = 0; z < 8; ++z) acc[i][j][z] = 0.0f;

#pragma unroll 2
  for (int k0 = 0; k0 < K; k0 += 32) {
    v16bf a0 = load_frag(A, K, m0,      k0, lane);
    v16bf a1 = load_frag(A, K, m0 + 16, k0, lane);
    v16bf b[4];
#pragma unroll
    for (int j = 0; j < 4; ++j) b[j] = load_frag(Wt, K, n0 + j * 16, k0, lane);
#pragma unroll
    for (int j = 0; j < 4; ++j) {
      acc[0][j] = wmma_bf16(a0, b[j], acc[0][j]);
      acc[1][j] = wmma_bf16(a1, b[j], acc[1][j]);
    }
  }

  // C-layout (ISA 7.12.2): VGPR i holds row +i (lanes 0-15) / +i+8
  // (lanes 16-31), column = n-tile base + (lane & 15).
  int half = lane >> 4;
  int nn   = lane & 15;
#pragma unroll
  for (int im = 0; im < 2; ++im)
#pragma unroll
    for (int jn = 0; jn < 4; ++jn)
#pragma unroll
      for (int i = 0; i < 8; ++i) {
        int m = m0 + im * 16 + i + half * 8;
        int n = n0 + jn * 16 + nn;
        float v = acc[im][jn][i];
        if (MODE == EP_F32) {
          if (bias)  v += bias[n];
          if (resid) v += resid[(size_t)m * N + n];
          outF[(size_t)m * N + n] = v;
        } else if (MODE == EP_GELU_BF16) {
          v += bias[n];
          v = 0.5f * v * (1.0f + erff(v * 0.7071067811865475f));
          outB[(size_t)m * N + n] = __float2bfloat16(v);
        } else {
          int b = m >> 11, s = m & (S_ - 1);     // S_ = 2048
          int h = n >> 6,  d = n & (DH_ - 1);
          if (MODE == EP_Q)
            outB[(((size_t)b * H_ + h) * S_ + s) * DH_ + d] =
                __float2bfloat16(v * 0.125f);    // fold 1/sqrt(dh)
          else if (MODE == EP_K)
            outB[(((size_t)b * H_ + h) * S_ + s) * DH_ + d] =
                __float2bfloat16(v);
          else  // EP_VT
            outB[(((size_t)b * H_ + h) * DH_ + d) * (size_t)S_ + s] =
                __float2bfloat16(v);
        }
      }
}

// ---------------------------------------------------------------------------
// Flash attention: grid (S/64, B*H), 128 threads = 4 waves, 16 q-rows/wave.
// Q fragments hoisted out of the kv loop (kv-invariant). Online softmax done
// directly in the WMMA C-layout; P staged via LDS (s_wait_dscnt fence) to
// feed the P@V WMMAs as A-fragments.
// ---------------------------------------------------------------------------
__global__ __launch_bounds__(128, 1) void flash_attn_kernel(
    const bf16_t* __restrict__ Q, const bf16_t* __restrict__ Kt,
    const bf16_t* __restrict__ Vt, const int* __restrict__ amask,
    bf16_t* __restrict__ ctx) {
  __shared__ __align__(16) bf16_t pbuf[4][16][64];
  int lane = threadIdx.x & 31;
  int wid  = threadIdx.x >> 5;
  int bh = blockIdx.y;
  int b  = bh >> 4;          // H_ = 16
  int h  = bh & 15;
  int qbase = blockIdx.x * 64 + wid * 16;
  const bf16_t* Qp = Q  + (size_t)bh * S_ * DH_;   // [S, dh]
  const bf16_t* Kp = Kt + (size_t)bh * S_ * DH_;   // [S, dh]
  const bf16_t* Vp = Vt + (size_t)bh * DH_ * S_;   // [dh, S]

  int half = lane >> 4;
  int nn   = lane & 15;

  // Q fragments are kv-invariant: load once.
  v16bf qf[2];
  qf[0] = load_frag(Qp, DH_, qbase, 0,  lane);
  qf[1] = load_frag(Qp, DH_, qbase, 32, lane);

  v8f o[4];
  float rmax[8], rsum[8];
#pragma unroll
  for (int i = 0; i < 8; ++i) { rmax[i] = -3.0e38f; rsum[i] = 0.0f; }
#pragma unroll
  for (int t = 0; t < 4; ++t)
#pragma unroll
    for (int i = 0; i < 8; ++i) o[t][i] = 0.0f;

  int kvend = (blockIdx.x + 1) * 64;               // causal upper bound
  for (int kv0 = 0; kv0 < kvend; kv0 += 64) {
    // ---- scores S = Q K^T (Q already scaled by 1/sqrt(dh)) ----
    v8f s[4];
#pragma unroll
    for (int t = 0; t < 4; ++t)
#pragma unroll
      for (int i = 0; i < 8; ++i) s[t][i] = 0.0f;
#pragma unroll
    for (int ks = 0; ks < 2; ++ks) {
#pragma unroll
      for (int t = 0; t < 4; ++t) {
        v16bf bk = load_frag(Kp, DH_, kv0 + t * 16, ks * 32, lane);
        s[t] = wmma_bf16(qf[ks], bk, s[t]);
      }
    }
    // ---- causal + attention_mask, per-chunk row max ----
    float cmax[8];
#pragma unroll
    for (int i = 0; i < 8; ++i) cmax[i] = -3.0e38f;
#pragma unroll
    for (int t = 0; t < 4; ++t) {
      int n = kv0 + t * 16 + nn;
      float km = (amask[b * S_ + n] != 0) ? 0.0f : -1.0e30f;
#pragma unroll
      for (int i = 0; i < 8; ++i) {
        int q = qbase + i + half * 8;
        float sv = s[t][i];
        sv = (n > q) ? -1.0e30f : (sv + km);
        s[t][i] = sv;
        cmax[i] = fmaxf(cmax[i], sv);
      }
    }
    // butterfly max across the 16-lane half that owns each row
#pragma unroll
    for (int i = 0; i < 8; ++i) {
#pragma unroll
      for (int mk = 8; mk >= 1; mk >>= 1)
        cmax[i] = fmaxf(cmax[i], __shfl_xor(cmax[i], mk, 32));
      float nm = fmaxf(rmax[i], cmax[i]);
      float sc = __expf(rmax[i] - nm);
      rmax[i] = nm;
      rsum[i] *= sc;
#pragma unroll
      for (int t = 0; t < 4; ++t) o[t][i] *= sc;
    }
    // ---- P = exp(S - max), stage to LDS as bf16 [16 x 64] ----
#pragma unroll
    for (int t = 0; t < 4; ++t)
#pragma unroll
      for (int i = 0; i < 8; ++i) {
        float p = __expf(s[t][i] - rmax[i]);
        rsum[i] += p;
        pbuf[wid][i + half * 8][t * 16 + nn] = __float2bfloat16(p);
      }
    asm volatile("s_wait_dscnt 0" ::: "memory");   // LDS store->load fence
    // ---- O += P @ V ----
#pragma unroll
    for (int k0 = 0; k0 < 64; k0 += 32) {
      v16bf ap = load_frag(&pbuf[wid][0][0], 64, 0, k0, lane);
#pragma unroll
      for (int t = 0; t < 4; ++t) {
        v16bf bv = load_frag(Vp, S_, t * 16, kv0 + k0, lane);
        o[t] = wmma_bf16(ap, bv, o[t]);
      }
    }
  }
  // ---- normalize and write ctx as [B,S,H*dh] bf16 (GEMM-ready) ----
#pragma unroll
  for (int t = 0; t < 4; ++t)
#pragma unroll
    for (int i = 0; i < 8; ++i) {
      int q = qbase + i + half * 8;
      int d = t * 16 + nn;
      float v = o[t][i] / rsum[i];
      ctx[((size_t)(b * S_ + q)) * D_ + h * DH_ + d] = __float2bfloat16(v);
    }
}

// ---------------------------------------------------------------------------
// Orchestration.
// ---------------------------------------------------------------------------
extern "C" void kernel_launch(void* const* d_in, const int* in_sizes, int n_in,
                              void* d_out, int out_size, void* d_ws,
                              size_t ws_size, hipStream_t stream) {
  (void)in_sizes; (void)n_in; (void)out_size; (void)ws_size;

  const float* x    = (const float*)d_in[0];
  const int*   am   = (const int*)d_in[1];
  const float* ln1a = (const float*)d_in[2];
  const float* ln1b = (const float*)d_in[3];
  const float* ln2a = (const float*)d_in[4];
  const float* ln2b = (const float*)d_in[5];
  const float* Wq   = (const float*)d_in[6];
  const float* Wk   = (const float*)d_in[7];
  const float* Wv   = (const float*)d_in[8];
  const float* Wo   = (const float*)d_in[9];
  const float* W1   = (const float*)d_in[10];
  const float* b1   = (const float*)d_in[11];
  const float* W2   = (const float*)d_in[12];
  const float* b2   = (const float*)d_in[13];
  float* out = (float*)d_out;

  // Workspace layout (~152 MiB total).
  char* ws = (char*)d_ws;
  const size_t MB = 1ull << 20;
  bf16_t* Wqt  = (bf16_t*)(ws + 0 * MB);     // 2 MiB  [D,D]^T
  bf16_t* Wkt  = (bf16_t*)(ws + 2 * MB);     // 2 MiB
  bf16_t* Wvt  = (bf16_t*)(ws + 4 * MB);     // 2 MiB
  bf16_t* Wot  = (bf16_t*)(ws + 6 * MB);     // 2 MiB
  bf16_t* W1t  = (bf16_t*)(ws + 8 * MB);     // 8 MiB  [Dff,D]
  bf16_t* W2t  = (bf16_t*)(ws + 16 * MB);    // 8 MiB  [D,Dff]
  bf16_t* ybf  = (bf16_t*)(ws + 24 * MB);    // 16 MiB [M,D]  (LN1 then LN2)
  float*  x1   = (float*) (ws + 40 * MB);    // 32 MiB [M,D]
  bf16_t* ctxb = (bf16_t*)(ws + 72 * MB);    // 16 MiB [M,D]
  bf16_t* Qb   = (bf16_t*)(ws + 88 * MB);    // 16 MiB [B,H,S,dh]
  bf16_t* Kb   = (bf16_t*)(ws + 104 * MB);   // 16 MiB [B,H,S,dh]
  bf16_t* Vtb  = (bf16_t*)(ws + 120 * MB);   // 16 MiB [B,H,dh,S]
  bf16_t* hb   = (bf16_t*)(ws + 88 * MB);    // 64 MiB [M,Dff] (reuses QKV)

  // 1) weights -> bf16, transposed to [N,K]
  cvt_t_kernel<<<(D_ * D_) / 256, 256, 0, stream>>>(Wq, Wqt, D_, D_);
  cvt_t_kernel<<<(D_ * D_) / 256, 256, 0, stream>>>(Wk, Wkt, D_, D_);
  cvt_t_kernel<<<(D_ * D_) / 256, 256, 0, stream>>>(Wv, Wvt, D_, D_);
  cvt_t_kernel<<<(D_ * D_) / 256, 256, 0, stream>>>(Wo, Wot, D_, D_);
  cvt_t_kernel<<<(D_ * DFF_) / 256, 256, 0, stream>>>(W1, W1t, D_, DFF_);
  cvt_t_kernel<<<(DFF_ * D_) / 256, 256, 0, stream>>>(W2, W2t, DFF_, D_);

  // 2) LN1
  ln_bf16_kernel<<<M_, 256, 0, stream>>>(x, ln1a, ln1b, ybf);

  // 3) QKV projections (Q folded with 1/sqrt(dh); V stored transposed)
  dim3 gD(M_ / 128, D_ / 128);
  gemm_wmma_kernel<EP_Q><<<gD, 256, 0, stream>>>(ybf, Wqt, M_, D_, D_,
                                                 nullptr, nullptr, nullptr, Qb);
  gemm_wmma_kernel<EP_K><<<gD, 256, 0, stream>>>(ybf, Wkt, M_, D_, D_,
                                                 nullptr, nullptr, nullptr, Kb);
  gemm_wmma_kernel<EP_VT><<<gD, 256, 0, stream>>>(ybf, Wvt, M_, D_, D_,
                                                  nullptr, nullptr, nullptr, Vtb);

  // 4) causal flash attention
  flash_attn_kernel<<<dim3(S_ / 64, B_ * H_), 128, 0, stream>>>(Qb, Kb, Vtb,
                                                                am, ctxb);

  // 5) output projection + residual -> x1 (f32)
  gemm_wmma_kernel<EP_F32><<<gD, 256, 0, stream>>>(ctxb, Wot, M_, D_, D_,
                                                   nullptr, x, x1, nullptr);

  // 6) LN2
  ln_bf16_kernel<<<M_, 256, 0, stream>>>(x1, ln2a, ln2b, ybf);

  // 7) FFN up + exact GELU -> h (bf16)
  gemm_wmma_kernel<EP_GELU_BF16><<<dim3(M_ / 128, DFF_ / 128), 256, 0, stream>>>(
      ybf, W1t, M_, DFF_, D_, b1, nullptr, nullptr, hb);

  // 8) FFN down + bias + residual -> d_out (f32)
  gemm_wmma_kernel<EP_F32><<<gD, 256, 0, stream>>>(hb, W2t, M_, D_, DFF_,
                                                   b2, x1, out, nullptr);
}
